// EdgePredictionHead_49031346651535
// MI455X (gfx1250) — matrix operand order repaired and checked by reading, not compiled, not run on hardware
//
#include <hip/hip_runtime.h>
#include <hip/hip_bf16.h>

typedef __attribute__((ext_vector_type(16))) __bf16 v16bf;
typedef __attribute__((ext_vector_type(2)))  __bf16 v2bf;
typedef __attribute__((ext_vector_type(2)))  float  v2f;
typedef __attribute__((ext_vector_type(8)))  float  v8f;

#define HID   128
#define NEDGE 400000
#define KCH   12      // 12 k-chunks of 32 -> K = 384 (GEMM1)
#define NT    4       // 4 n-tiles of 16  -> cols 0..63 (valid 0..49)
#define EPW   16      // edges per wave
#define WPB   8       // waves per block
#define SHSTR 76      // LDS row stride for h1 (16B aligned, conflict-free)
#define PW1_DW (KCH * NT * 32 * 8)   // 12288 dwords
#define PW2_DW (2 * 2 * 32 * 8)      // 1024 dwords

// f32 pair -> packed bf16 dword (lowers to v_cvt_pk_bf16_f32 on gfx1250)
__device__ __forceinline__ unsigned int pkbf(float lo, float hi) {
    v2f f = {lo, hi};
    v2bf b = __builtin_convertvector(f, v2bf);
    unsigned int u;
    __builtin_memcpy(&u, &b, 4);
    return u;
}

// single-instruction ReLU: v_med3_f32 x, 0, +inf
__device__ __forceinline__ float relu1(float v) {
    return __builtin_amdgcn_fmed3f(v, 0.f, __builtin_inff());
}

// ---------------------------------------------------------------------------
// Prepack W1 [384,50] and W2 [50,25] (f32) into bf16 B-operand 32x16 tiles.
// Within a tile: lane l holds col N=l%16, h=l/16, dword d holds
// (K0=8h+2d+8*(d/4), K0+1). Out-of-range rows/cols zero-padded.
// pw layout: [0, PW1_DW) = W1 tiles (kc*NT+nt); [PW1_DW, +PW2_DW) = W2 tiles.
// ---------------------------------------------------------------------------
__global__ void pack_w(const float* __restrict__ W1, const float* __restrict__ W2,
                       unsigned int* __restrict__ pw) {
    int gid = blockIdx.x * 256 + threadIdx.x;
    if (gid >= PW1_DW + PW2_DW) return;
    if (gid < PW1_DW) {
        int d    = gid & 7;
        int lane = (gid >> 3) & 31;
        int tile = gid >> 8;
        int kc = tile / NT, nt = tile % NT;
        int N   = lane & 15;
        int K0  = ((lane >> 4) << 3) + 2 * d + ((d >> 2) << 3);
        int col = nt * 16 + N;
        int r0  = kc * 32 + K0;
        float v0 = (col < 50) ? W1[r0 * 50 + col]       : 0.f;
        float v1 = (col < 50) ? W1[(r0 + 1) * 50 + col] : 0.f;
        pw[gid] = pkbf(v0, v1);
    } else {
        int g2   = gid - PW1_DW;
        int d    = g2 & 7;
        int lane = (g2 >> 3) & 31;
        int tile = g2 >> 8;                  // 0..3
        int kc = tile >> 1, nt = tile & 1;
        int N   = lane & 15;
        int K0  = ((lane >> 4) << 3) + 2 * d + ((d >> 2) << 3);
        int col = nt * 16 + N;
        int r0  = kc * 32 + K0;
        float v0 = (r0     < 50 && col < 25) ? W2[r0 * 25 + col]       : 0.f;
        float v1 = (r0 + 1 < 50 && col < 25) ? W2[(r0 + 1) * 25 + col] : 0.f;
        pw[gid] = pkbf(v0, v1);
    }
}

// ---------------------------------------------------------------------------
// Fused edge MLP. GEMM1: gather+relu+concat folded into K (12x4 WMMA).
// GEMM2+layer3: WMMA on LDS-staged h1, W3 folded into a shfl_xor reduction.
// ---------------------------------------------------------------------------
__global__ __launch_bounds__(256) void edge_mlp(
    const float* __restrict__ x, const float* __restrict__ ea,
    const int* __restrict__ ei, const unsigned int* __restrict__ pw,
    const float* __restrict__ b1, const float* __restrict__ b2,
    const float* __restrict__ W3, const float* __restrict__ b3,
    float* __restrict__ out)
{
    __shared__ __align__(16) float sH[WPB][16][SHSTR];  // h1 tiles, f32
    __shared__ float sB1[64];
    __shared__ float sB2[32], sW3[32];                  // zero-padded to 32

    int t = threadIdx.x;
    if (t < 64) sB1[t] = (t < 50) ? b1[t] : 0.f;
    if (t < 32) { sB2[t] = (t < 25) ? b2[t] : 0.f; sW3[t] = (t < 25) ? W3[t] : 0.f; }
    __syncthreads();

    int wave = t >> 5, lane = t & 31;
    int row = lane & 15;                       // edge row within tile
    int khalf = lane >> 4;
    int e0 = (blockIdx.x * WPB + wave) * EPW;
    int e = e0 + row;
    int srcN = ei[e];
    int dstN = ei[NEDGE + e];

    // ---------------- GEMM1: [16,384] @ [384,64] ----------------
    v8f c[NT];
#pragma unroll
    for (int nt = 0; nt < NT; nt++) c[nt] = (v8f){0, 0, 0, 0, 0, 0, 0, 0};

#pragma unroll
    for (int kc = 0; kc < KCH; kc++) {
        const float* src;
        bool doRelu;
        if (kc < 4)      { src = x  + (size_t)srcN * HID + kc * 32;       doRelu = true;  }
        else if (kc < 8) { src = x  + (size_t)dstN * HID + (kc - 4) * 32; doRelu = true;  }
        else             { src = ea + (size_t)e    * HID + (kc - 8) * 32; doRelu = false; }

        // A lane layout (16-bit A 16x32): lane<16 holds K {0..7,16..23},
        // lane>=16 holds K {8..15,24..31} of its row.
        const float4* p0 = (const float4*)(src + khalf * 8);
        const float4* p1 = (const float4*)(src + 16 + khalf * 8);
        float4 f0 = p0[0], f1 = p0[1], f2 = p1[0], f3 = p1[1];
        if (doRelu) {
            f0.x = relu1(f0.x); f0.y = relu1(f0.y); f0.z = relu1(f0.z); f0.w = relu1(f0.w);
            f1.x = relu1(f1.x); f1.y = relu1(f1.y); f1.z = relu1(f1.z); f1.w = relu1(f1.w);
            f2.x = relu1(f2.x); f2.y = relu1(f2.y); f2.z = relu1(f2.z); f2.w = relu1(f2.w);
            f3.x = relu1(f3.x); f3.y = relu1(f3.y); f3.z = relu1(f3.z); f3.w = relu1(f3.w);
        }
        union { unsigned int u[8]; v16bf v; } A;
        A.u[0] = pkbf(f0.x, f0.y); A.u[1] = pkbf(f0.z, f0.w);
        A.u[2] = pkbf(f1.x, f1.y); A.u[3] = pkbf(f1.z, f1.w);
        A.u[4] = pkbf(f2.x, f2.y); A.u[5] = pkbf(f2.z, f2.w);
        A.u[6] = pkbf(f3.x, f3.y); A.u[7] = pkbf(f3.z, f3.w);

#pragma unroll
        for (int nt = 0; nt < NT; nt++) {
            const uint4* bp = (const uint4*)(pw + ((size_t)(kc * NT + nt) * 32 + lane) * 8);
            union { uint4 q[2]; v16bf v; } B;
            B.q[0] = bp[0]; B.q[1] = bp[1];
            c[nt] = __builtin_amdgcn_wmma_f32_16x16x32_bf16(
                false, A.v, false, B.v, (short)0, c[nt], false, false);
        }
    }

    // bias + relu in D layout, spill h1 to LDS (cols 50..63 become exact zeros)
#pragma unroll
    for (int nt = 0; nt < NT; nt++) {
        int col = nt * 16 + (lane & 15);
        float bb = sB1[col];
#pragma unroll
        for (int r = 0; r < 8; r++) {
            sH[wave][r + khalf * 8][col] = relu1(c[nt][r] + bb);
        }
    }
    __syncthreads();

    // ---------------- GEMM2: [16,64] @ [64,32] via WMMA ----------------
    const unsigned int* pw2 = pw + PW1_DW;
    v8f c2[2];
    c2[0] = (v8f){0, 0, 0, 0, 0, 0, 0, 0};
    c2[1] = (v8f){0, 0, 0, 0, 0, 0, 0, 0};

#pragma unroll
    for (int kc = 0; kc < 2; kc++) {
        const float4* hA0 = (const float4*)(&sH[wave][row][kc * 32 + khalf * 8]);
        const float4* hA1 = (const float4*)(&sH[wave][row][kc * 32 + 16 + khalf * 8]);
        float4 f0 = hA0[0], f1 = hA0[1], f2 = hA1[0], f3 = hA1[1];
        union { unsigned int u[8]; v16bf v; } A;
        A.u[0] = pkbf(f0.x, f0.y); A.u[1] = pkbf(f0.z, f0.w);
        A.u[2] = pkbf(f1.x, f1.y); A.u[3] = pkbf(f1.z, f1.w);
        A.u[4] = pkbf(f2.x, f2.y); A.u[5] = pkbf(f2.z, f2.w);
        A.u[6] = pkbf(f3.x, f3.y); A.u[7] = pkbf(f3.z, f3.w);

#pragma unroll
        for (int nt = 0; nt < 2; nt++) {
            const uint4* bp = (const uint4*)(pw2 + ((size_t)(kc * 2 + nt) * 32 + lane) * 8);
            union { uint4 q[2]; v16bf v; } B;
            B.q[0] = bp[0]; B.q[1] = bp[1];
            c2[nt] = __builtin_amdgcn_wmma_f32_16x16x32_bf16(
                false, A.v, false, B.v, (short)0, c2[nt], false, false);
        }
    }

    // ---------------- bias2+relu, fold W3, reduce across columns ----------------
    // Lane owns col j (tile0: j=lane%16, tile1: j=16+lane%16). Padded j>=25 have
    // b2=0, W2col=0, W3=0 -> contribute exactly 0.
    float b3v = b3[0];
    int jc = lane & 15;
    float w30 = sW3[jc], w31 = sW3[16 + jc];
    float bb0 = sB2[jc], bb1 = sB2[16 + jc];
    float g[8];
#pragma unroll
    for (int r = 0; r < 8; r++) {
        float g0 = relu1(c2[0][r] + bb0) * w30;
        float g1 = relu1(c2[1][r] + bb1) * w31;
        g[r] = g0 + g1;
    }
    // butterfly sum over the 16 columns (stays within each 16-lane half)
#pragma unroll
    for (int m = 1; m < 16; m <<= 1) {
#pragma unroll
        for (int r = 0; r < 8; r++)
            g[r] += __shfl_xor(g[r], m, 32);
    }
    // lane 0 holds rows 0..7, lane 16 holds rows 8..15
    if ((lane & 15) == 0) {
        int rbase = khalf * 8;
#pragma unroll
        for (int r = 0; r < 8; r++)
            out[e0 + rbase + r] = g[r] + b3v;
    }
}

extern "C" void kernel_launch(void* const* d_in, const int* in_sizes, int n_in,
                              void* d_out, int out_size, void* d_ws, size_t ws_size,
                              hipStream_t stream) {
    const float* x  = (const float*)d_in[0];
    const float* ea = (const float*)d_in[1];
    const int*   ei = (const int*)d_in[2];
    const float* W1 = (const float*)d_in[3];
    const float* b1 = (const float*)d_in[4];
    const float* W2 = (const float*)d_in[5];
    const float* b2 = (const float*)d_in[6];
    const float* W3 = (const float*)d_in[7];
    const float* b3 = (const float*)d_in[8];
    float* out = (float*)d_out;
    unsigned int* pw = (unsigned int*)d_ws;    // 13312 dwords = 52 KB

    const int packDwords = PW1_DW + PW2_DW;
    pack_w<<<(packDwords + 255) / 256, 256, 0, stream>>>(W1, W2, pw);

    const int blocks = NEDGE / (EPW * WPB);    // 3125 exactly
    edge_mlp<<<blocks, 256, 0, stream>>>(x, ea, ei, pw, b1, b2, W3, b3, out);
}
